// SequenceMemoryUpdater_36979668419203
// MI455X (gfx1250) — matrix operand order repaired and checked by reading, not compiled, not run on hardware
//
#include <hip/hip_runtime.h>
#include <hip/hip_bf16.h>
#include <stdint.h>

// ---------------------------------------------------------------------------
// TGN memory updater for MI455X (gfx1250, wave32, WMMA)
//   out_memory = memory; out_memory[ids] = GRUCell(messages, memory[ids])
//   out_last_update = last_update; out_last_update[ids] = timestamps
// ---------------------------------------------------------------------------

#define N_NODES 1000000
#define MEM_DIM 128
#define MSG_DIM 256

typedef __attribute__((ext_vector_type(16))) __bf16        v16bf;
typedef __attribute__((ext_vector_type(8)))  float         v8f;
typedef __attribute__((ext_vector_type(4)))  unsigned int  u32x4;
typedef __attribute__((ext_vector_type(4)))  float         f32x4;

union FragBF {
    v16bf bf;
    u32x4 q[2];
};

__device__ __forceinline__ unsigned short f2bf(float f) {
    unsigned int u = __float_as_uint(f);
    u += 0x7FFFu + ((u >> 16) & 1u);   // round-to-nearest-even
    return (unsigned short)(u >> 16);
}

__device__ __forceinline__ v8f wmma_bf16(const FragBF& a, const FragBF& b, v8f c) {
    return __builtin_amdgcn_wmma_f32_16x16x32_bf16(
        /*neg_a=*/false, a.bf, /*neg_b=*/false, b.bf,
        /*c_mod=*/(short)0, c, /*reuse_a=*/false, /*reuse_b=*/false);
}

__device__ __forceinline__ float sigmoidf_(float x) {
    return 1.0f / (1.0f + __expf(-x));
}

// ---------------------------------------------------------------------------
// Kernel 1: bulk copy memory -> out[0:128M], last_update -> out[128M:129M].
// Pure streaming: use non-temporal load+store so the 1 GB stream does not
// evict the packed weights / gathered rows that the GRU kernel keeps in L2.
// ---------------------------------------------------------------------------
__global__ void copy_kernel(const f32x4* __restrict__ mem4,
                            const f32x4* __restrict__ lu4,
                            f32x4* __restrict__ out4,
                            long n_mem4, long n_tot4) {
    long i      = (long)blockIdx.x * blockDim.x + threadIdx.x;
    long stride = (long)gridDim.x * blockDim.x;
    for (; i < n_tot4; i += stride) {
        f32x4 v = (i < n_mem4) ? __builtin_nontemporal_load(&mem4[i])
                               : __builtin_nontemporal_load(&lu4[i - n_mem4]);
        __builtin_nontemporal_store(v, &out4[i]);
    }
}

// ---------------------------------------------------------------------------
// Kernel 2: pack W_ih / W_hh (f32, row-major [384 x K]) into bf16 B-fragment
// layout: fragment (t, ks) is 32x16 (K x N) with lane l holding 16 contiguous
// bf16: N = l&15, K = (l<16 ? 0 : 16) + p, stored lane-major (32B per lane).
//   W_ih frags at elem 0      : ((t*8 + ks)*32 + l)*16 + p   (24*8 frags)
//   W_hh frags at elem 98304  : ((t*4 + ks)*32 + l)*16 + p   (24*4 frags)
// ---------------------------------------------------------------------------
__global__ void pack_weights(const float* __restrict__ Wih,
                             const float* __restrict__ Whh,
                             unsigned short* __restrict__ dst) {
    const int NIH = 24 * 8 * 512;   // 98304
    const int NHH = 24 * 4 * 512;   // 49152
    int tid = blockIdx.x * blockDim.x + threadIdx.x;
    if (tid < NIH) {
        int p  = tid & 15;
        int l  = (tid >> 4) & 31;
        int ks = (tid >> 9) & 7;
        int t  = tid >> 12;
        int j  = t * 16 + (l & 15);
        int k  = ks * 32 + ((l < 16) ? 0 : 16) + p;
        dst[tid] = f2bf(Wih[j * MSG_DIM + k]);
    } else if (tid < NIH + NHH) {
        int idx = tid - NIH;
        int p  = idx & 15;
        int l  = (idx >> 4) & 31;
        int ks = (idx >> 9) & 3;
        int t  = idx >> 11;
        int j  = t * 16 + (l & 15);
        int k  = ks * 32 + ((l < 16) ? 0 : 16) + p;
        dst[tid] = f2bf(Whh[j * MEM_DIM + k]);
    }
}

// ---------------------------------------------------------------------------
// Kernel 3: fused gather + GRUCell (bf16 WMMA, f32 accumulate) + scatter.
// One block = 32 rows (2 M-tiles of 16), 8 waves. Wave w owns N-tiles
// {2w, 2w+1} (r/z: both GEMMs summed) and tile 16+w (n gate: i_n, h_n kept
// separate). Gate exchange through LDS planes.
// ---------------------------------------------------------------------------
__global__ __launch_bounds__(256)
void gru_kernel(const float* __restrict__ memory,
                const int*   __restrict__ ids,
                const float* __restrict__ messages,
                const float* __restrict__ ts,
                const float* __restrict__ b_ih,
                const float* __restrict__ b_hh,
                const unsigned short* __restrict__ wpk,
                float* __restrict__ out_mem,
                float* __restrict__ out_lu) {
    // LDS layout (40.1 KB):
    //   [0,16384)         xs  : 32x256 bf16  (aliased as rplane 32x128 f32 later)
    //   [16384,24576)     hs  : 32x128 bf16
    //   [24576,40960)     nplane : 32x128 f32
    //   [40960,41088)     sid : 32 int
    __shared__ __align__(16) unsigned char smem[16384 + 8192 + 16384 + 128];
    unsigned short* xs     = (unsigned short*)smem;
    float*          rplane = (float*)smem;
    unsigned short* hs     = (unsigned short*)(smem + 16384);
    float*          nplane = (float*)(smem + 16384 + 8192);
    int*            sid    = (int*)(smem + 16384 + 8192 + 16384);

    const int tid = threadIdx.x;
    const int l   = tid & 31;
    const int w   = tid >> 5;
    const int r0  = blockIdx.x * 32;

    if (tid < 32) sid[tid] = ids[r0 + tid];
    __syncthreads();

    // ---- stage x rows (messages) as bf16 ----
    for (int i = tid; i < 2048; i += 256) {          // 32 rows * 64 float4
        int row = i >> 6, cp = i & 63;
        f32x4 v = ((const f32x4*)messages)[(long)(r0 + row) * 64 + cp];
        unsigned int p0 = (unsigned int)f2bf(v.x) | ((unsigned int)f2bf(v.y) << 16);
        unsigned int p1 = (unsigned int)f2bf(v.z) | ((unsigned int)f2bf(v.w) << 16);
        ((unsigned int*)xs)[row * 128 + cp * 2]     = p0;
        ((unsigned int*)xs)[row * 128 + cp * 2 + 1] = p1;
    }
    // ---- stage h rows (gathered memory) as bf16 ----
    for (int i = tid; i < 1024; i += 256) {          // 32 rows * 32 float4
        int row = i >> 5, cp = i & 31;
        f32x4 v = ((const f32x4*)memory)[(long)sid[row] * 32 + cp];
        unsigned int p0 = (unsigned int)f2bf(v.x) | ((unsigned int)f2bf(v.y) << 16);
        unsigned int p1 = (unsigned int)f2bf(v.z) | ((unsigned int)f2bf(v.w) << 16);
        ((unsigned int*)hs)[row * 64 + cp * 2]     = p0;
        ((unsigned int*)hs)[row * 64 + cp * 2 + 1] = p1;
    }
    __syncthreads();

    // ---- WMMA GEMMs ----
    v8f acc_rz[2][2] = {};   // [mt][ti]  tiles 2w, 2w+1  (gi+gh summed)
    v8f acc_in[2]    = {};   // [mt]      tile 16+w, x-side (i_n)
    v8f acc_hn[2]    = {};   // [mt]      tile 16+w, h-side (h_n)

    const int kb_a   = (l < 16) ? 0 : 16;  // A-fragment lane K base
    const int kb_row = l & 15;
    const unsigned short* wih = wpk;
    const unsigned short* whh = wpk + 98304;
    const int t0 = 2 * w, t1 = 2 * w + 1, tn = 16 + w;

    // x @ W_ih^T  (K = 256, 8 k-steps)
    for (int ks = 0; ks < 8; ++ks) {
        const int k0 = ks * 32;
        FragBF a[2], b0, b1, bn;
        for (int mt = 0; mt < 2; ++mt) {
            int base = (mt * 16 + kb_row) * 256 + k0 + kb_a;
            a[mt].q[0] = *(const u32x4*)&xs[base];
            a[mt].q[1] = *(const u32x4*)&xs[base + 8];
        }
        { const unsigned short* p = wih + ((t0 * 8 + ks) * 32 + l) * 16;
          b0.q[0] = *(const u32x4*)p; b0.q[1] = *(const u32x4*)(p + 8); }
        { const unsigned short* p = wih + ((t1 * 8 + ks) * 32 + l) * 16;
          b1.q[0] = *(const u32x4*)p; b1.q[1] = *(const u32x4*)(p + 8); }
        { const unsigned short* p = wih + ((tn * 8 + ks) * 32 + l) * 16;
          bn.q[0] = *(const u32x4*)p; bn.q[1] = *(const u32x4*)(p + 8); }
        for (int mt = 0; mt < 2; ++mt) {
            acc_rz[mt][0] = wmma_bf16(a[mt], b0, acc_rz[mt][0]);
            acc_rz[mt][1] = wmma_bf16(a[mt], b1, acc_rz[mt][1]);
            acc_in[mt]    = wmma_bf16(a[mt], bn, acc_in[mt]);
        }
    }
    // h @ W_hh^T  (K = 128, 4 k-steps)
    for (int ks = 0; ks < 4; ++ks) {
        const int k0 = ks * 32;
        FragBF a[2], b0, b1, bn;
        for (int mt = 0; mt < 2; ++mt) {
            int base = (mt * 16 + kb_row) * 128 + k0 + kb_a;
            a[mt].q[0] = *(const u32x4*)&hs[base];
            a[mt].q[1] = *(const u32x4*)&hs[base + 8];
        }
        { const unsigned short* p = whh + ((t0 * 4 + ks) * 32 + l) * 16;
          b0.q[0] = *(const u32x4*)p; b0.q[1] = *(const u32x4*)(p + 8); }
        { const unsigned short* p = whh + ((t1 * 4 + ks) * 32 + l) * 16;
          b1.q[0] = *(const u32x4*)p; b1.q[1] = *(const u32x4*)(p + 8); }
        { const unsigned short* p = whh + ((tn * 4 + ks) * 32 + l) * 16;
          bn.q[0] = *(const u32x4*)p; bn.q[1] = *(const u32x4*)(p + 8); }
        for (int mt = 0; mt < 2; ++mt) {
            acc_rz[mt][0] = wmma_bf16(a[mt], b0, acc_rz[mt][0]);
            acc_rz[mt][1] = wmma_bf16(a[mt], b1, acc_rz[mt][1]);
            acc_hn[mt]    = wmma_bf16(a[mt], bn, acc_hn[mt]);
        }
    }
    __syncthreads();   // all A reads done -> safe to overwrite xs with rplane

    // ---- r gate (waves 0..3 own tiles j in [0,128)) ----
    if (w < 4) {
        for (int ti = 0; ti < 2; ++ti) {
            int j = (2 * w + ti) * 16 + kb_row;          // r-dim d = j
            float bias = b_ih[j] + b_hh[j];
            for (int mt = 0; mt < 2; ++mt)
                for (int g = 0; g < 8; ++g) {
                    int row = mt * 16 + g + ((l < 16) ? 0 : 8);
                    rplane[row * 128 + j] = sigmoidf_(acc_rz[mt][ti][g] + bias);
                }
        }
    }
    __syncthreads();

    // ---- n gate: every wave owns n-dims [16w, 16w+16) ----
    {
        int nd = 16 * w + kb_row;
        int j  = 256 + nd;
        float bi = b_ih[j], bh = b_hh[j];
        for (int mt = 0; mt < 2; ++mt)
            for (int g = 0; g < 8; ++g) {
                int row = mt * 16 + g + ((l < 16) ? 0 : 8);
                float r = rplane[row * 128 + nd];
                nplane[row * 128 + nd] =
                    tanhf(acc_in[mt][g] + bi + r * (acc_hn[mt][g] + bh));
            }
    }
    __syncthreads();

    // ---- z gate + combine + scatter (waves 4..7 own z dims) ----
    if (w >= 4) {
        int q = w - 4;
        for (int ti = 0; ti < 2; ++ti) {
            int j  = (8 + 2 * q + ti) * 16 + kb_row;      // 128..255
            int zd = j - 128;
            float bias = b_ih[j] + b_hh[j];
            for (int mt = 0; mt < 2; ++mt)
                for (int g = 0; g < 8; ++g) {
                    int row = mt * 16 + g + ((l < 16) ? 0 : 8);
                    float z = sigmoidf_(acc_rz[mt][ti][g] + bias);
                    float n = nplane[row * 128 + zd];
                    long  gr = (long)sid[row];
                    float h = memory[gr * 128 + zd];      // f32 h for z*h term
                    float o = (1.0f - z) * n + z * h;
                    // updated rows are not re-read: bypass cache residency
                    __builtin_nontemporal_store(o, &out_mem[gr * 128 + zd]);
                }
        }
    }
    // timestamps scatter
    if (tid < 32) out_lu[sid[tid]] = ts[r0 + tid];
}

// ---------------------------------------------------------------------------
extern "C" void kernel_launch(void* const* d_in, const int* in_sizes, int n_in,
                              void* d_out, int out_size, void* d_ws, size_t ws_size,
                              hipStream_t stream) {
    const float* memory   = (const float*)d_in[0];
    const float* last_up  = (const float*)d_in[1];
    const int*   ids      = (const int*)d_in[2];
    const float* messages = (const float*)d_in[3];
    const float* ts       = (const float*)d_in[4];
    const float* W_ih     = (const float*)d_in[5];
    const float* W_hh     = (const float*)d_in[6];
    const float* b_ih     = (const float*)d_in[7];
    const float* b_hh     = (const float*)d_in[8];
    float* out = (float*)d_out;
    unsigned short* wpk = (unsigned short*)d_ws;   // 294,912 bytes used

    const long n_mem4 = (long)N_NODES * MEM_DIM / 4;   // 32,000,000
    const long n_tot4 = n_mem4 + N_NODES / 4;          // 32,250,000
    copy_kernel<<<4096, 256, 0, stream>>>((const f32x4*)memory,
                                          (const f32x4*)last_up,
                                          (f32x4*)out, n_mem4, n_tot4);

    pack_weights<<<576, 256, 0, stream>>>(W_ih, W_hh, wpk);

    const int U = in_sizes[2];          // 200,000 (divisible by 32)
    const int nblocks = U / 32;         // 6,250
    gru_kernel<<<nblocks, 256, 0, stream>>>(memory, ids, messages, ts,
                                            b_ih, b_hh, wpk,
                                            out, out + (long)N_NODES * MEM_DIM);
}
